// GNN_34291018891310
// MI455X (gfx1250) — compile-verified, matrix operand order checked by reading, not run on hardware
//
#include <hip/hip_runtime.h>
#include <hip/hip_bf16.h>

typedef __attribute__((ext_vector_type(16))) _Float16 v16h;
typedef __attribute__((ext_vector_type(8)))  float    v8f;

#define NTHREADS 256

static __device__ __forceinline__ float leaky(float v) {
    return v > 0.0f ? v : 0.2f * v;
}

// ---------------------------------------------------------------------------
// Degree / normalization:  deg[i] = 1 (self loop) + indegree(i);  dis = rsqrt
// ---------------------------------------------------------------------------
__global__ void deg_init(float* __restrict__ deg, int n) {
    int i = blockIdx.x * blockDim.x + threadIdx.x;
    if (i < n) deg[i] = 1.0f;   // self loop contribution
}

__global__ void deg_accum(const int* __restrict__ dst, float* __restrict__ deg, int e) {
    int i = blockIdx.x * blockDim.x + threadIdx.x;
    if (i < e) atomicAdd(&deg[dst[i]], 1.0f);
}

__global__ void deg_to_dis(float* __restrict__ deg, int n) {
    int i = blockIdx.x * blockDim.x + threadIdx.x;
    if (i < n) deg[i] = rsqrtf(deg[i]);   // deg >= 1 always
}

// ---------------------------------------------------------------------------
// Swizzle W1 (384x32 f32, row-major) into WMMA B-fragment order, f16.
// Fragment f = kt*2 + nt (kt in [0,12), nt in {0,1}); per lane 16 halves.
// B element (K,N): lane = 16*(K in upper half8 group) ... per ISA layout:
//   lane L: Ncol = L%16, h = L/16;  element e: K = kt*32 + 16*(e/8) + 8*h + e%8
// ---------------------------------------------------------------------------
__global__ void swizzle_w1(const float* __restrict__ W1, _Float16* __restrict__ w1sw) {
    int idx = blockIdx.x * blockDim.x + threadIdx.x;      // 0 .. 24*32*16-1
    if (idx >= 24 * 32 * 16) return;
    int e    = idx & 15;
    int lane = (idx >> 4) & 31;
    int f    = idx >> 9;
    int kt   = f >> 1;
    int nt   = f & 1;
    int ncol = lane & 15;
    int h    = lane >> 4;
    int K    = kt * 32 + 16 * (e >> 3) + 8 * h + (e & 7);
    w1sw[idx] = (_Float16)W1[K * 32 + nt * 16 + ncol];
}

// ---------------------------------------------------------------------------
// GEMM1: h1[N,32] = x[N,384] @ W1[384,32]  via v_wmma_f32_16x16x32_f16.
// One wave per 16-row M-tile; two accumulators cover the 32 output columns.
// ---------------------------------------------------------------------------
__global__ void __launch_bounds__(NTHREADS) gemm1_wmma(const float* __restrict__ x,
                                                       const _Float16* __restrict__ w1sw,
                                                       float* __restrict__ h1) {
    const int wave = threadIdx.x >> 5;
    const int lane = threadIdx.x & 31;
    const int m0   = (blockIdx.x * 8 + wave) * 16;
    const int ncol = lane & 15;
    const int hh   = lane >> 4;

    const float* __restrict__ arow = x + (size_t)(m0 + ncol) * 384 + 8 * hh;
    const v16h* __restrict__ bfr   = (const v16h*)w1sw;

    v8f acc0 = {};
    v8f acc1 = {};

#pragma unroll
    for (int kt = 0; kt < 12; ++kt) {
        // A fragment: two contiguous 32B chunks per lane (K = 8h+e, K = 16+8h+e)
        const float* ap = arow + kt * 32;
        float4 c0 = *(const float4*)(ap + 0);
        float4 c1 = *(const float4*)(ap + 4);
        float4 c2 = *(const float4*)(ap + 16);
        float4 c3 = *(const float4*)(ap + 20);
        v16h a;
        a[0]  = (_Float16)c0.x; a[1]  = (_Float16)c0.y; a[2]  = (_Float16)c0.z; a[3]  = (_Float16)c0.w;
        a[4]  = (_Float16)c1.x; a[5]  = (_Float16)c1.y; a[6]  = (_Float16)c1.z; a[7]  = (_Float16)c1.w;
        a[8]  = (_Float16)c2.x; a[9]  = (_Float16)c2.y; a[10] = (_Float16)c2.z; a[11] = (_Float16)c2.w;
        a[12] = (_Float16)c3.x; a[13] = (_Float16)c3.y; a[14] = (_Float16)c3.z; a[15] = (_Float16)c3.w;

        v16h b0 = bfr[(kt * 2 + 0) * 32 + lane];
        v16h b1 = bfr[(kt * 2 + 1) * 32 + lane];

        acc0 = __builtin_amdgcn_wmma_f32_16x16x32_f16(false, a, false, b0,
                                                      (short)0, acc0, false, false);
        acc1 = __builtin_amdgcn_wmma_f32_16x16x32_f16(false, a, false, b1,
                                                      (short)0, acc1, false, false);
    }

    // C/D layout: lane -> N = lane%16 (+16 for acc1), VGPR r -> M = r + 8*(lane/16)
#pragma unroll
    for (int r = 0; r < 8; ++r) {
        int m = m0 + r + 8 * hh;
        h1[(size_t)m * 32 + ncol]      = acc0[r];
        h1[(size_t)m * 32 + 16 + ncol] = acc1[r];
    }
}

// ---------------------------------------------------------------------------
// Edge aggregation: agg[dst, f] += h[src, f] * dis[src]*dis[dst]
// over E edges plus N self loops. One lane per (edge, feature).
// ---------------------------------------------------------------------------
template <int F>
__global__ void edge_agg(const int* __restrict__ src, const int* __restrict__ dst,
                         const float* __restrict__ dis, const float* __restrict__ hin,
                         float* __restrict__ agg, int E, int N) {
    int gid  = blockIdx.x * blockDim.x + threadIdx.x;
    int edge = gid / F;
    int f    = gid & (F - 1);
    if (edge >= E + N) return;
    int s, d;
    if (edge < E) { s = src[edge]; d = dst[edge]; }
    else          { s = d = edge - E; }
    float w = dis[s] * dis[d];
    atomicAdd(&agg[(size_t)d * F + f], hin[(size_t)s * F + f] * w);
}

// ---------------------------------------------------------------------------
// Layer-2 projection: h2[i,0:8] = leaky(agg1[i,:] + b1) @ W2   (32 -> 8)
// ---------------------------------------------------------------------------
__global__ void proj2(const float* __restrict__ agg1, const float* __restrict__ b1,
                      const float* __restrict__ W2, float* __restrict__ h2, int n) {
    int i = blockIdx.x * blockDim.x + threadIdx.x;
    if (i >= n) return;
    float acc[8] = {0.f, 0.f, 0.f, 0.f, 0.f, 0.f, 0.f, 0.f};
    const float* r = agg1 + (size_t)i * 32;
#pragma unroll
    for (int j = 0; j < 32; ++j) {
        float v = leaky(r[j] + b1[j]);
#pragma unroll
        for (int o = 0; o < 8; ++o) acc[o] += v * W2[j * 8 + o];
    }
#pragma unroll
    for (int o = 0; o < 8; ++o) h2[(size_t)i * 8 + o] = acc[o];
}

// ---------------------------------------------------------------------------
// Per-graph mean readout + MLP head. One block per graph (NPG nodes).
// ---------------------------------------------------------------------------
__global__ void __launch_bounds__(NTHREADS) readout_head(
        const float* __restrict__ agg2, const float* __restrict__ b2,
        const float* __restrict__ W3, const float* __restrict__ b3,
        const float* __restrict__ W4, const float* __restrict__ b4,
        float* __restrict__ out, int npg) {
    __shared__ float red[NTHREADS * 8];
    const int b = blockIdx.x;
    const int t = threadIdx.x;

    float p[8] = {0.f, 0.f, 0.f, 0.f, 0.f, 0.f, 0.f, 0.f};
    for (int n = t; n < npg; n += NTHREADS) {
        const float* r = agg2 + ((size_t)b * npg + n) * 8;
#pragma unroll
        for (int j = 0; j < 8; ++j) p[j] += leaky(r[j] + b2[j]);
    }
#pragma unroll
    for (int j = 0; j < 8; ++j) red[t * 8 + j] = p[j];
    __syncthreads();

    for (int s = NTHREADS / 2; s > 0; s >>= 1) {
        if (t < s) {
#pragma unroll
            for (int j = 0; j < 8; ++j) red[t * 8 + j] += red[(t + s) * 8 + j];
        }
        __syncthreads();
    }

    if (t == 0) {
        float g[8], q[8];
        float inv = 1.0f / (float)npg;
#pragma unroll
        for (int j = 0; j < 8; ++j) g[j] = red[j] * inv;
#pragma unroll
        for (int k = 0; k < 8; ++k) {
            float v = b3[k];
#pragma unroll
            for (int j = 0; j < 8; ++j) v += g[j] * W3[j * 8 + k];
            q[k] = leaky(v);
        }
#pragma unroll
        for (int c = 0; c < 2; ++c) {
            float v = b4[c];
#pragma unroll
            for (int k = 0; k < 8; ++k) v += q[k] * W4[k * 2 + c];
            out[b * 2 + c] = v;
        }
    }
}

// ---------------------------------------------------------------------------
// Host-side orchestration (graph-capture safe: only stream ops)
// ---------------------------------------------------------------------------
extern "C" void kernel_launch(void* const* d_in, const int* in_sizes, int n_in,
                              void* d_out, int out_size, void* d_ws, size_t ws_size,
                              hipStream_t stream) {
    const float* x   = (const float*)d_in[0];
    const int*   ei  = (const int*)d_in[1];      // [2, E]
    // d_in[2] = batch (unused: graphs are NPG consecutive nodes)
    const float* W1  = (const float*)d_in[3];
    const float* b1  = (const float*)d_in[4];
    const float* W2  = (const float*)d_in[5];
    const float* b2  = (const float*)d_in[6];
    const float* W3  = (const float*)d_in[7];
    const float* b3  = (const float*)d_in[8];
    const float* W4  = (const float*)d_in[9];
    const float* b4  = (const float*)d_in[10];
    float*       out = (float*)d_out;

    const int N   = in_sizes[0] / 384;   // 262144
    const int E   = in_sizes[1] / 2;     // 4194304
    const int B   = out_size / 2;        // 64
    const int NPG = N / B;               // 4096

    const int* src = ei;
    const int* dst = ei + E;

    // Workspace carve-up (all 256B aligned)
    char* ws = (char*)d_ws;
    float*    dis  = (float*)ws;            ws += (size_t)N * 4;            // 1 MB
    float*    bufA = (float*)ws;            ws += (size_t)N * 32 * 4;       // h1, later h2proj
    float*    bufB = (float*)ws;            ws += (size_t)N * 32 * 4;       // agg1, later agg2
    _Float16* w1sw = (_Float16*)ws;         ws += 24 * 32 * 16 * 2;         // 24 KB

    float* h1    = bufA;
    float* agg1  = bufB;
    float* h2p   = bufA;   // alias: h1 dead once proj2 reads agg1
    float* agg2  = bufB;   // alias: agg1 dead after proj2

    // Normalization: dis = rsqrt(1 + indegree)
    deg_init <<<(N + NTHREADS - 1) / NTHREADS, NTHREADS, 0, stream>>>(dis, N);
    deg_accum<<<(E + NTHREADS - 1) / NTHREADS, NTHREADS, 0, stream>>>(dst, dis, E);
    deg_to_dis<<<(N + NTHREADS - 1) / NTHREADS, NTHREADS, 0, stream>>>(dis, N);

    // GEMM1 via WMMA
    swizzle_w1<<<(24 * 32 * 16 + NTHREADS - 1) / NTHREADS, NTHREADS, 0, stream>>>(W1, w1sw);
    gemm1_wmma<<<N / (16 * 8), NTHREADS, 0, stream>>>(x, w1sw, h1);

    // Layer-1 aggregation (agg1 = scatter-add of h1[src]*norm)
    hipMemsetAsync(agg1, 0, (size_t)N * 32 * 4, stream);
    {
        long long tot = (long long)(E + N) * 32;
        edge_agg<32><<<(unsigned)((tot + NTHREADS - 1) / NTHREADS), NTHREADS, 0, stream>>>(
            src, dst, dis, h1, agg1, E, N);
    }

    // Layer-2 projection (fuses bias + leaky of layer 1)
    proj2<<<(N + NTHREADS - 1) / NTHREADS, NTHREADS, 0, stream>>>(agg1, b1, W2, h2p, N);

    // Layer-2 aggregation
    hipMemsetAsync(agg2, 0, (size_t)N * 8 * 4, stream);
    {
        long long tot = (long long)(E + N) * 8;
        edge_agg<8><<<(unsigned)((tot + NTHREADS - 1) / NTHREADS), NTHREADS, 0, stream>>>(
            src, dst, dis, h2p, agg2, E, N);
    }

    // Mean readout + head
    readout_head<<<B, NTHREADS, 0, stream>>>(agg2, b2, W3, b3, W4, b4, out, NPG);

    (void)n_in; (void)ws_size; (void)ws;
}
